// SS_Conv_SSM_4733053960424
// MI455X (gfx1250) — matrix operand order repaired and checked by reading, not compiled, not run on hardware
//
#include <hip/hip_runtime.h>
#include <hip/hip_bf16.h>

// ---------------------------------------------------------------------------
// CDNA5 (gfx1250) implementation of the SS-Conv-SSM block.
// All dense contractions use v_wmma_f32_16x16x32_f16 (f16 in, f32 accum).
// LDS tiles are staged in *fragment order* so each lane loads its whole
// 32-byte WMMA fragment with vector ds_load_b128s (no scalar gather).
// ---------------------------------------------------------------------------

typedef __attribute__((ext_vector_type(16))) _Float16 v16h;
typedef __attribute__((ext_vector_type(8)))  float    v8f;

#define M_TOT   16384   // B*H*W = 4*64*64
#define BATCH   4

__device__ __forceinline__ float siluf(float v) { return v / (1.f + __expf(-v)); }

// Inverse fragment maps (CDNA5 ISA §7.12.2), k in [0,32):
// A 16x32 f16:  lane = hi*16 + m
__device__ __forceinline__ void a_slot(int kr, int& hi, int& i) {
  hi = (kr >> 3) & 1;
  i  = (kr & 7) | ((kr >> 4) << 3);
}
// B 32x16 f16:  lane = hi*16 + n (same k order as A)
__device__ __forceinline__ void b_slot(int kr, int& hi, int& i) {
  hi = (kr >> 3) & 1;
  i  = (kr & 7) | ((kr >> 4) << 3);
}

enum { EPI_NONE = 0, EPI_SILU_BN = 1, EPI_SILU_BN_BN = 2, EPI_RELU_BIAS = 3, EPI_SHUFFLE_RES = 4 };

// ---------------------------------------------------------------------------
// Generic 1x1-conv GEMM:  out[pos, n] = epi( sum_k A[pos, a_off+k] * W[n, k] )
// Block = 128 threads (4 waves); block owns a 16-row M tile; wave w owns
// n-tiles w, w+4, ...   A fragment hoisted across n-tiles (kk-outer loop).
// ---------------------------------------------------------------------------
template <int K, int N, int EPI>
__launch_bounds__(128)
__global__ void gemm1x1(const float* __restrict__ A, int a_stride, int a_off,
                        const float* __restrict__ W,
                        const float* __restrict__ g1, const float* __restrict__ b1,
                        const float* __restrict__ g2, const float* __restrict__ b2,
                        const float* __restrict__ resid,
                        float* __restrict__ out, int o_stride, int o_off)
{
  constexpr int KC = K / 32;          // k-chunks
  constexpr int NT = N / 16;          // n-tiles
  constexpr int MAXT = (NT + 3) / 4;  // n-tiles per wave (max)
  __shared__ v16h AsF[KC][32];        // fragment-order A tile
  __shared__ v16h BsF[NT][KC][32];    // fragment-order W tiles
  const int tid = threadIdx.x;
  const int m0  = blockIdx.x * 16;

  // stage A (float4 global loads, scatter to fragment slots)
  for (int j = tid; j < 16 * K / 4; j += 128) {
    int m = j / (K / 4), k4 = (j % (K / 4)) * 4;
    const float4 v4 = *(const float4*)&A[(size_t)(m0 + m) * a_stride + a_off + k4];
    const float vv[4] = {v4.x, v4.y, v4.z, v4.w};
#pragma unroll
    for (int e = 0; e < 4; ++e) {
      int k = k4 + e, hi, i;
      a_slot(k & 31, hi, i);
      ((_Float16*)&AsF[k >> 5][hi * 16 + m])[i] = (_Float16)vv[e];
    }
  }
  // stage W: W is (N, K) row-major -> coalesced float4 over k
  for (int j = tid; j < N * K / 4; j += 128) {
    int n = j / (K / 4), k4 = (j % (K / 4)) * 4;
    const float4 v4 = *(const float4*)&W[(size_t)n * K + k4];
    const float vv[4] = {v4.x, v4.y, v4.z, v4.w};
#pragma unroll
    for (int e = 0; e < 4; ++e) {
      int k = k4 + e, hi, i;
      b_slot(k & 31, hi, i);
      ((_Float16*)&BsF[n >> 4][k >> 5][hi * 16 + (n & 15)])[i] = (_Float16)vv[e];
    }
  }
  __syncthreads();

  const int lane = tid & 31;
  const int wave = tid >> 5;
  const int hi   = lane >> 4;
  const int mn   = lane & 15;

  v8f acc[MAXT];
#pragma unroll
  for (int t = 0; t < MAXT; ++t) acc[t] = (v8f){0.f, 0.f, 0.f, 0.f, 0.f, 0.f, 0.f, 0.f};

#pragma unroll
  for (int kc = 0; kc < KC; ++kc) {
    const v16h af = AsF[kc][lane];
#pragma unroll
    for (int t = 0; t < MAXT; ++t) {
      const int nt = wave + t * 4;
      if (nt < NT) {                       // wave-uniform: EXEC stays full
        const v16h bf = BsF[nt][kc][lane];
        acc[t] = __builtin_amdgcn_wmma_f32_16x16x32_f16(false, af, false, bf,
                                                        (short)0, acc[t], false, false);
      }
    }
  }

#pragma unroll
  for (int t = 0; t < MAXT; ++t) {
    const int nt = wave + t * 4;
    if (nt >= NT) continue;
#pragma unroll
    for (int v = 0; v < 8; ++v) {
      int m = v + (hi << 3);
      int n = nt * 16 + mn;
      size_t pos = (size_t)(m0 + m);
      float y = acc[t][v];
      if constexpr (EPI == EPI_NONE) {
        out[pos * o_stride + o_off + n] = y;
      } else if constexpr (EPI == EPI_SILU_BN) {
        out[pos * o_stride + o_off + n] = siluf(y * g1[n] + b1[n]);
      } else if constexpr (EPI == EPI_SILU_BN_BN) {
        out[pos * o_stride + o_off + n] = siluf(y * g1[n] + b1[n]) * g2[n] + b2[n];
      } else if constexpr (EPI == EPI_RELU_BIAS) {
        float r = y + b1[n];
        out[pos * o_stride + o_off + n] = r > 0.f ? r : 0.f;
      } else if constexpr (EPI == EPI_SHUFFLE_RES) {
        // channel shuffle: reshape(2,64).swap -> j = 2*(n&63) + (n>>6); +residual
        int j = ((n & 63) << 1) + (n >> 6);
        out[pos * o_stride + j] = siluf(y * g1[n] + b1[n]) + resid[pos * 128 + j];
      }
    }
  }
}

// ---------------------------------------------------------------------------
// 3x3 conv 64->64 (SAME) as implicit GEMM: 9 shifted 16x64 A tiles, WMMA
// accumulate across taps.  Epilogue: relu((y + cb)*g + b).
// blockIdx.x = ((b*64 + h)*4 + wt); tile = 16 consecutive w positions.
// ---------------------------------------------------------------------------
__launch_bounds__(128)
__global__ void conv3x3_bnrelu(const float* __restrict__ A,   // (B,64,64,64)
                               const float* __restrict__ W,   // (3,3,64,64) HWIO
                               const float* __restrict__ cb,
                               const float* __restrict__ g, const float* __restrict__ bb,
                               float* __restrict__ out)
{
  __shared__ v16h AsF[2][32];      // fragment-order A tile (K=64 -> 2 chunks)
  __shared__ v16h BsF[4][2][32];   // fragment-order W tiles (N=64 -> 4 ntiles)
  const int tid  = threadIdx.x;
  const int wt   = blockIdx.x & 3;
  const int h    = (blockIdx.x >> 2) & 63;
  const int b    = blockIdx.x >> 8;
  const int lane = tid & 31;
  const int wave = tid >> 5;
  const int hi   = lane >> 4;
  const int mn   = lane & 15;

  v8f acc = {0.f, 0.f, 0.f, 0.f, 0.f, 0.f, 0.f, 0.f};

  for (int t = 0; t < 9; ++t) {
    const int dy = t / 3 - 1, dx = t % 3 - 1;
    __syncthreads();
    // A tile: rows = 16 shifted positions, cols = 64 in-channels
    for (int j = tid; j < 16 * 16; j += 128) {     // 16 rows x 16 float4
      int m = j >> 4, k4 = (j & 15) * 4;
      int hh = h + dy, ww = wt * 16 + m + dx;
      float4 v4 = {0.f, 0.f, 0.f, 0.f};
      if (hh >= 0 && hh < 64 && ww >= 0 && ww < 64)
        v4 = *(const float4*)&A[(((size_t)b * 64 + hh) * 64 + ww) * 64 + k4];
      const float vv[4] = {v4.x, v4.y, v4.z, v4.w};
#pragma unroll
      for (int e = 0; e < 4; ++e) {
        int k = k4 + e, fhi, fi;
        a_slot(k & 31, fhi, fi);
        ((_Float16*)&AsF[k >> 5][fhi * 16 + m])[fi] = (_Float16)vv[e];
      }
    }
    // W tile for this tap: (ci, co); stage transposed into fragment order
    for (int j = tid; j < 64 * 16; j += 128) {     // 64 ci x 16 float4 of co
      int k = j >> 4, n4 = (j & 15) * 4;           // k = ci, n = co
      const float4 v4 = *(const float4*)&W[((size_t)t * 64 + k) * 64 + n4];
      const float vv[4] = {v4.x, v4.y, v4.z, v4.w};
      int fhi, fi;
      b_slot(k & 31, fhi, fi);
#pragma unroll
      for (int e = 0; e < 4; ++e) {
        int n = n4 + e;
        ((_Float16*)&BsF[n >> 4][k >> 5][fhi * 16 + (n & 15)])[fi] = (_Float16)vv[e];
      }
    }
    __syncthreads();
#pragma unroll
    for (int kc = 0; kc < 2; ++kc) {
      const v16h af = AsF[kc][lane];
      const v16h bf = BsF[wave][kc][lane];
      acc = __builtin_amdgcn_wmma_f32_16x16x32_f16(false, af, false, bf,
                                                   (short)0, acc, false, false);
    }
  }
#pragma unroll
  for (int v = 0; v < 8; ++v) {
    int m = v + (hi << 3);
    int n = wave * 16 + mn;
    size_t pos = ((size_t)b * 64 + h) * 64 + wt * 16 + m;
    float y = (acc[v] + cb[n]) * g[n] + bb[n];
    out[pos * 64 + n] = y > 0.f ? y : 0.f;
  }
}

// ---------------------------------------------------------------------------
// Depthwise 3x3 + bias + SiLU on xh (= xz channels 0..127, stride 256)
// ---------------------------------------------------------------------------
__global__ void dwconv_silu(const float* __restrict__ xz, const float* __restrict__ w,
                            const float* __restrict__ bias, float* __restrict__ xhc)
{
  int idx = blockIdx.x * blockDim.x + threadIdx.x;   // M_TOT*128 threads
  int c = idx & 127;
  int p = idx >> 7;
  int wcol = p & 63, hrow = (p >> 6) & 63, b = p >> 12;
  float s = bias[c];
#pragma unroll
  for (int t = 0; t < 9; ++t) {
    int hh = hrow + t / 3 - 1, ww = wcol + t % 3 - 1;
    if (hh < 0 || hh >= 64 || ww < 0 || ww >= 64) continue;
    s += xz[(((size_t)b * 64 + hh) * 64 + ww) * 256 + c] * w[t * 128 + c];
  }
  xhc[(size_t)p * 128 + c] = siluf(s);
}

// ---------------------------------------------------------------------------
// Selective scan: one block per (b, direction); thread d owns one channel's
// 16-wide recurrent state in registers.  y written at the *spatial* position
// each direction visits, so the 4 direction buffers sum directly.
// ---------------------------------------------------------------------------
__device__ __forceinline__ int scan_pos(int k, int l) {
  if (k == 0) return l;
  if (k == 1) return ((l & 63) << 6) | (l >> 6);
  if (k == 2) return 4095 - l;
  int lp = 4095 - l;
  return ((lp & 63) << 6) | (lp >> 6);
}

__launch_bounds__(128)
__global__ void ss_scan(const float* __restrict__ dbl, const float* __restrict__ xhc,
                        const float* __restrict__ A_logs, const float* __restrict__ dt_w,
                        const float* __restrict__ dt_b, const float* __restrict__ Ds,
                        float* __restrict__ ybuf)
{
  const int k = blockIdx.x & 3;
  const int b = blockIdx.x >> 2;
  const int d = threadIdx.x;
  const int kd = k * 128 + d;

  float Ar[16];
#pragma unroll
  for (int s = 0; s < 16; ++s) Ar[s] = -__expf(A_logs[kd * 16 + s]);
  const float w0 = dt_w[kd * 4 + 0], w1 = dt_w[kd * 4 + 1];
  const float w2 = dt_w[kd * 4 + 2], w3 = dt_w[kd * 4 + 3];
  const float tb = dt_b[kd];
  const float Dd = Ds[kd];
  float hst[16];
#pragma unroll
  for (int s = 0; s < 16; ++s) hst[s] = 0.f;

  const size_t base = (size_t)b * 4096;
  for (int l = 0; l < 4096; ++l) {
    const int sp = scan_pos(k, l);
    const float* dp = dbl + (base + sp) * 144 + k * 36;
    const float  u  = xhc[(base + sp) * 128 + d];

    // prefetch a few steps ahead (emits global_prefetch_b8 on gfx1250)
    int ln = l + 4; if (ln > 4095) ln = 4095;
    const int spn = scan_pos(k, ln);
    __builtin_prefetch(dbl + (base + spn) * 144 + k * 36, 0, 3);
    __builtin_prefetch(xhc + (base + spn) * 128 + d, 0, 3);

    float dv = tb + w0 * dp[0] + w1 * dp[1] + w2 * dp[2] + w3 * dp[3];
    float delta = (dv > 20.f) ? dv : log1pf(__expf(dv));
    float du = delta * u;
    float y = 0.f;
#pragma unroll
    for (int s = 0; s < 16; ++s) {
      float dA = __expf(delta * Ar[s]);
      hst[s] = dA * hst[s] + du * dp[4 + s];
      y += hst[s] * dp[20 + s];
    }
    ybuf[((base + sp) * 4 + k) * 128 + d] = y + Dd * u;
  }
}

// ---------------------------------------------------------------------------
// Sum 4 directions + LayerNorm(128) + gate with silu(z)
// ---------------------------------------------------------------------------
__launch_bounds__(128)
__global__ void combine_ln(const float* __restrict__ ybuf, const float* __restrict__ xz,
                           const float* __restrict__ lng, const float* __restrict__ lnb,
                           float* __restrict__ yg)
{
  __shared__ float s1[128], s2[128];
  const size_t pos = blockIdx.x;
  const int d = threadIdx.x;
  const float* yp = ybuf + pos * 512 + d;
  float v = yp[0] + yp[128] + yp[256] + yp[384];
  s1[d] = v; s2[d] = v * v;
  __syncthreads();
  for (int s = 64; s > 0; s >>= 1) {
    if (d < s) { s1[d] += s1[d + s]; s2[d] += s2[d + s]; }
    __syncthreads();
  }
  float mean = s1[0] * (1.f / 128.f);
  float var  = s2[0] * (1.f / 128.f) - mean * mean;
  float nv = (v - mean) * rsqrtf(var + 1e-5f) * lng[d] + lnb[d];
  float z  = xz[pos * 256 + 128 + d];
  yg[pos * 128 + d] = nv * siluf(z);
}

// ---------------------------------------------------------------------------
extern "C" void kernel_launch(void* const* d_in, const int* in_sizes, int n_in,
                              void* d_out, int out_size, void* d_ws, size_t ws_size,
                              hipStream_t stream)
{
  const float* x         = (const float*)d_in[0];
  const float* cv1_w     = (const float*)d_in[1];
  const float* cv1_g     = (const float*)d_in[2];
  const float* cv1_b     = (const float*)d_in[3];
  const float* in_proj_w = (const float*)d_in[4];
  const float* dw_w      = (const float*)d_in[5];
  const float* dw_b      = (const float*)d_in[6];
  const float* x_proj_w  = (const float*)d_in[7];
  const float* dt_w      = (const float*)d_in[8];
  const float* dt_b      = (const float*)d_in[9];
  const float* A_logs    = (const float*)d_in[10];
  const float* Ds        = (const float*)d_in[11];
  const float* ln_g      = (const float*)d_in[12];
  const float* ln_b      = (const float*)d_in[13];
  const float* out_proj_w= (const float*)d_in[14];
  const float* cv2_w     = (const float*)d_in[15];
  const float* cv2_g     = (const float*)d_in[16];
  const float* cv2_b     = (const float*)d_in[17];
  const float* bn0_g     = (const float*)d_in[18];
  const float* bn0_b     = (const float*)d_in[19];
  const float* c1_w      = (const float*)d_in[20];
  const float* c1_b      = (const float*)d_in[21];
  const float* bn1_g     = (const float*)d_in[22];
  const float* bn1_b     = (const float*)d_in[23];
  const float* c2_w      = (const float*)d_in[24];
  const float* c2_b      = (const float*)d_in[25];
  const float* bn2_g     = (const float*)d_in[26];
  const float* bn2_b     = (const float*)d_in[27];
  const float* c3_w      = (const float*)d_in[28];
  const float* c3_b      = (const float*)d_in[29];
  const float* cv_w      = (const float*)d_in[30];
  const float* cv_g      = (const float*)d_in[31];
  const float* cv_b      = (const float*)d_in[32];
  const float* cv3_w     = (const float*)d_in[33];
  const float* cv3_g     = (const float*)d_in[34];
  const float* cv3_b     = (const float*)d_in[35];

  float* ws = (float*)d_ws;
  size_t off = 0;
  float* rbuf = ws + off; off += (size_t)M_TOT * 64;    // silu(bn(cv1(right)))
  float* xz   = ws + off; off += (size_t)M_TOT * 256;   // in_proj out (xh | z)
  float* xhc  = ws + off; off += (size_t)M_TOT * 128;   // silu(dwconv(xh))
  float* dbl  = ws + off; off += (size_t)M_TOT * 144;   // x_proj, 4 dirs * 36
  float* ybuf = ws + off; off += (size_t)M_TOT * 512;   // per-direction scan y
  float* yg   = ws + off; off += (size_t)M_TOT * 128;   // LN-gated
  float* xr   = ws + off; off += (size_t)M_TOT * 64;    // out_proj
  float* cat  = ws + off; off += (size_t)M_TOT * 128;   // [left | right] concat
  float* lA   = ws + off; off += (size_t)M_TOT * 64;
  float* lB   = ws + off; off += (size_t)M_TOT * 64;
  (void)ws_size; (void)in_sizes; (void)n_in; (void)out_size;

  dim3 blk(128);
  const int MG = M_TOT / 16;   // 1024 M tiles

  // right branch: cv1 (1x1) + bn + silu
  gemm1x1<64, 64, EPI_SILU_BN><<<MG, blk, 0, stream>>>(
      x, 128, 64, cv1_w, cv1_g, cv1_b, nullptr, nullptr, nullptr, rbuf, 64, 0);
  // in_proj 64 -> 256
  gemm1x1<64, 256, EPI_NONE><<<MG, blk, 0, stream>>>(
      rbuf, 64, 0, in_proj_w, nullptr, nullptr, nullptr, nullptr, nullptr, xz, 256, 0);
  // depthwise 3x3 + silu
  dwconv_silu<<<(M_TOT * 128) / 256, 256, 0, stream>>>(xz, dw_w, dw_b, xhc);
  // x_proj for all 4 directions: 128 -> 144 (= 9 exact 16-wide tiles)
  gemm1x1<128, 144, EPI_NONE><<<MG, blk, 0, stream>>>(
      xhc, 128, 0, x_proj_w, nullptr, nullptr, nullptr, nullptr, nullptr, dbl, 144, 0);
  // selective scan, 16 blocks = (b, k)
  ss_scan<<<BATCH * 4, 128, 0, stream>>>(dbl, xhc, A_logs, dt_w, dt_b, Ds, ybuf);
  // direction sum + LN + silu(z) gate
  combine_ln<<<M_TOT, 128, 0, stream>>>(ybuf, xz, ln_g, ln_b, yg);
  // out_proj 128 -> 64
  gemm1x1<128, 64, EPI_NONE><<<MG, blk, 0, stream>>>(
      yg, 128, 0, out_proj_w, nullptr, nullptr, nullptr, nullptr, nullptr, xr, 64, 0);
  // cv (1x1) + bn + silu -> right half of concat
  gemm1x1<64, 64, EPI_SILU_BN><<<MG, blk, 0, stream>>>(
      xr, 64, 0, cv_w, cv_g, cv_b, nullptr, nullptr, nullptr, cat, 128, 64);
  // left branch: cv2 + bn + silu, then bn0
  gemm1x1<64, 64, EPI_SILU_BN_BN><<<MG, blk, 0, stream>>>(
      x, 128, 0, cv2_w, cv2_g, cv2_b, bn0_g, bn0_b, nullptr, lA, 64, 0);
  conv3x3_bnrelu<<<MG, blk, 0, stream>>>(lA, c1_w, c1_b, bn1_g, bn1_b, lB);
  conv3x3_bnrelu<<<MG, blk, 0, stream>>>(lB, c2_w, c2_b, bn2_g, bn2_b, lA);
  // c3 (1x1) + bias + relu -> left half of concat
  gemm1x1<64, 64, EPI_RELU_BIAS><<<MG, blk, 0, stream>>>(
      lA, 64, 0, c3_w, nullptr, c3_b, nullptr, nullptr, nullptr, cat, 128, 0);
  // cv3 (1x1, 128->128) + bn + silu + channel shuffle + residual
  gemm1x1<128, 128, EPI_SHUFFLE_RES><<<MG, blk, 0, stream>>>(
      cat, 128, 0, cv3_w, cv3_g, cv3_b, nullptr, nullptr, x, (float*)d_out, 128, 0);
}
